// SingleFrameFlowTransformer_18236431139347
// MI455X (gfx1250) — matrix operand order repaired and checked by reading, not compiled, not run on hardware
//
#include <hip/hip_runtime.h>
#include <hip/hip_bf16.h>
#include <math.h>

// ---------------------------------------------------------------------------
// SingleFrameFlowTransformer forward for MI455X (gfx1250, wave32, WMMA).
// Big GEMMs (notably the 65536x512x512 cross-attn K/V projections, ~412 GFLOP
// total) run through v_wmma_f32_16x16x32_f16 with f16 weights pre-transposed
// into workspace (L2-resident). GEMM A-strip is block-staged in LDS (f16,
// XOR-swizzled, conflict-free); B-fragments use a copy-free ping-pong
// register pipeline so VMEM overlaps the WMMA stream. Everything else VALU.
// ---------------------------------------------------------------------------

typedef __attribute__((ext_vector_type(16))) _Float16 v16h;
typedef __attribute__((ext_vector_type(8)))  _Float16 v8h;
typedef __attribute__((ext_vector_type(8)))  float    v8f;

#define Bsz 32
#define Jn  27
#define Tt  (Bsz*Jn)      // 864 token rows
#define Dd  512
#define Nn  2048          // point cloud size
#define Hh  8
#define DH  64
#define FFd 2048
#define NEGBIG (-1.0e9f)

__constant__ int c_parent[Jn] = {0,0,1,2,3,4,5,6,7,8,8,3,11,12,13,14,15,15,0,18,19,20,0,22,23,24,3};

// ---------------- weight transpose + f32 -> f16 convert --------------------
// Wt[n*K + k] = (f16) W[k*N + n]
__global__ __launch_bounds__(256) void wconv(const float* __restrict__ w,
                                             _Float16* __restrict__ wt,
                                             int K, int N) {
    size_t idx = (size_t)blockIdx.x * 256 + threadIdx.x;
    if (idx >= (size_t)K * N) return;
    int k = (int)(idx % K);
    int n = (int)(idx / K);
    wt[idx] = (_Float16)w[(size_t)k * N + n];
}

// ---------------- WMMA GEMM: C[M,N] = A[M,K] @ W[K,N] (+bias,+res,relu) ----
// Wt is f16, transposed [N][K]. Requirements (all launches satisfy them):
//   M%16==0, N%64==0, K%64==0, tilesN=(N/64)%8==0, tiles%8==0 (exact grid).
// Block = 8 waves sharing one 16-row A strip (staged once in LDS as f16 with
// XOR chunk swizzle -> conflict-free ds_load_b128 fragment reads). Each wave
// owns a 16x64 strip of C. B fragments run a 2-phase ping-pong pipeline with
// read-then-overwrite (pure WAR) so no register copies are needed and each
// buffer's reload overlaps the other buffer's 4 WMMAs.
#define WMMA4(a, bb)                                                                               \
    acc[0] = __builtin_amdgcn_wmma_f32_16x16x32_f16(false, a, false, bb[0], (short)0, acc[0], false, false); \
    acc[1] = __builtin_amdgcn_wmma_f32_16x16x32_f16(false, a, false, bb[1], (short)0, acc[1], false, false); \
    acc[2] = __builtin_amdgcn_wmma_f32_16x16x32_f16(false, a, false, bb[2], (short)0, acc[2], false, false); \
    acc[3] = __builtin_amdgcn_wmma_f32_16x16x32_f16(false, a, false, bb[3], (short)0, acc[3], false, false);

template<bool A_F16, bool OUT_F16, bool RELU, bool RES>
__global__ __launch_bounds__(256) void gemm_wmma(const void* __restrict__ Av,
                                                 const _Float16* __restrict__ Wt,
                                                 const float* __restrict__ bias,
                                                 void* __restrict__ Cv,
                                                 const float* __restrict__ Res,
                                                 int M, int N, int K) {
    extern __shared__ _Float16 As[];               // 16*K halves, swizzled
    const int lane = threadIdx.x & 31;
    const int wave = threadIdx.x >> 5;
    const int tilesN = N >> 6;                     // 64-wide N tiles
    const int tm = (blockIdx.x * 8) / tilesN;      // uniform across the block
    const int tn = (blockIdx.x * 8 + wave) % tilesN;
    const int m0 = tm << 4, n0 = tn << 6;
    const int hi = lane >> 4;                      // half-wave selector
    const int rA = lane & 15;

    // ---- cooperative A-strip stage: global (f32|f16) -> LDS f16, swizzled.
    // element (r,k) stored at r*K + ((chunk ^ (r&15))<<3) + (k&7), chunk=k>>3
    const int CH = K >> 3;                         // 16-byte chunks per row
    for (int cid = threadIdx.x; cid < 16 * CH; cid += 256) {
        int r = cid / CH, c = cid % CH;
        v8h val;
        if constexpr (A_F16) {
            val = *(const v8h*)((const _Float16*)Av + (size_t)(m0 + r) * K + (c << 3));
        } else {
            v8f f = *(const v8f*)((const float*)Av + (size_t)(m0 + r) * K + (c << 3));
            #pragma unroll
            for (int e = 0; e < 8; ++e) val[e] = (_Float16)f[e];
        }
        *(v8h*)(As + (size_t)r * K + ((size_t)((c ^ (r & 15)) << 3))) = val;
    }
    __syncthreads();

    const _Float16* Arow = As + (size_t)rA * K;
    const _Float16* Wbase = Wt + (size_t)(n0 + rA) * K + (hi << 4);
    const size_t wstep = (size_t)16 * K;

    auto loadB = [&](v16h* dst, int kk) {
        const _Float16* Wp = Wbase + kk;
        dst[0] = *(const v16h*)(Wp);
        dst[1] = *(const v16h*)(Wp + wstep);
        dst[2] = *(const v16h*)(Wp + 2 * wstep);
        dst[3] = *(const v16h*)(Wp + 3 * wstep);
    };
    auto loadA = [&](int kk) -> v16h {
        const int c0 = (kk >> 3) + (hi << 1);      // low 8 = chunk c0, high = c0+1
        v8h alo = *(const v8h*)(Arow + ((c0 ^ rA) << 3));
        v8h ahi = *(const v8h*)(Arow + (((c0 + 1) ^ rA) << 3));
        v16h a;
        #pragma unroll
        for (int e = 0; e < 8; ++e) { a[e] = alo[e]; a[e + 8] = ahi[e]; }
        return a;
    };

    v8f acc[4] = {v8f{}, v8f{}, v8f{}, v8f{}};
    v16h bA[4], bB[4];                             // ping-pong B fragments
    loadB(bA, 0);
    loadB(bB, 32);
    int k0 = 0;
    for (; k0 + 64 < K; k0 += 64) {
        v16h a0 = loadA(k0);
        WMMA4(a0, bA);                             // reads bA ...
        loadB(bA, k0 + 64);                        // ... then reloads it (WAR)
        v16h a1 = loadA(k0 + 32);
        WMMA4(a1, bB);
        loadB(bB, k0 + 96);
    }
    {   // peeled tail: k0 == K-64
        v16h a0 = loadA(k0);
        WMMA4(a0, bA);
        v16h a1 = loadA(k0 + 32);
        WMMA4(a1, bB);
    }

    // ---- epilogue: C/D layout row = r + 8*hi, col = n0 + j*16 + lane&15
    #pragma unroll
    for (int j = 0; j < 4; ++j) {
        const int col = n0 + j * 16 + rA;
        const float bv = bias ? bias[col] : 0.f;
        #pragma unroll
        for (int r = 0; r < 8; ++r) {
            const int m = m0 + r + (hi << 3);
            float v = acc[j][r] + bv;
            if constexpr (RES)  v += Res[(size_t)m * N + col];
            if constexpr (RELU) v = v > 0.f ? v : 0.f;
            if constexpr (OUT_F16) ((_Float16*)Cv)[(size_t)m * N + col] = (_Float16)v;
            else                   ((float*)Cv)[(size_t)m * N + col] = v;
        }
    }
}

static inline int gemm_grid(int M, int N) { return (((M >> 4) * (N >> 6)) + 7) / 8; }
static inline unsigned gemm_lds(int K) { return (unsigned)(16 * K * 2); }

// ---------------- small-K embedding kernels --------------------------------
__global__ __launch_bounds__(256) void mem_embed(const float* __restrict__ pc,
                                                 const float* __restrict__ w,
                                                 const float* __restrict__ b,
                                                 _Float16* __restrict__ out) {
    size_t idx = (size_t)blockIdx.x * 256 + threadIdx.x;   // 65536*512
    if (idx >= (size_t)Bsz * Nn * Dd) return;
    int d = (int)(idx & (Dd - 1));
    size_t bn = idx >> 9;
    const float* p = pc + bn * 6;
    float acc = b[d];
    #pragma unroll
    for (int c = 0; c < 6; ++c) acc += p[c] * w[c * Dd + d];
    out[idx] = (_Float16)acc;
}

// h[t,d] = sum_{c<3} in[t*3+c]*w[c*Dd+d] + b[d]
__global__ __launch_bounds__(256) void proj3(const float* __restrict__ in,
                                             const float* __restrict__ w,
                                             const float* __restrict__ b,
                                             float* __restrict__ out) {
    size_t idx = (size_t)blockIdx.x * 256 + threadIdx.x;   // Tt*Dd
    if (idx >= (size_t)Tt * Dd) return;
    int d = (int)(idx & (Dd - 1));
    size_t t = idx >> 9;
    const float* p = in + t * 3;
    out[idx] = b[d] + p[0] * w[0 * Dd + d] + p[1] * w[1 * Dd + d] + p[2] * w[2 * Dd + d];
}

// cc[t, 0:512] = hxt[t]; cc[t, 512:1024] = hxt[(b, parent[j])]
__global__ __launch_bounds__(256) void concat_parent(const float* __restrict__ hxt,
                                                     float* __restrict__ cc) {
    size_t idx = (size_t)blockIdx.x * 256 + threadIdx.x;   // Tt*Dd
    if (idx >= (size_t)Tt * Dd) return;
    int d = (int)(idx & (Dd - 1));
    int t = (int)(idx >> 9);
    int b = t / Jn, j = t % Jn;
    float v0 = hxt[(size_t)t * Dd + d];
    float v1 = hxt[((size_t)(b * Jn + c_parent[j])) * Dd + d];
    cc[(size_t)t * 2 * Dd + d]      = v0;
    cc[(size_t)t * 2 * Dd + Dd + d] = v1;
}

// t1[b,d] = silu(tau[b]*t_w1[d] + t_b1[d])
__global__ __launch_bounds__(256) void tau_silu(const float* __restrict__ tau,
                                                const float* __restrict__ w1,
                                                const float* __restrict__ b1,
                                                float* __restrict__ out) {
    int idx = blockIdx.x * 256 + threadIdx.x;              // Bsz*Dd
    if (idx >= Bsz * Dd) return;
    int d = idx & (Dd - 1);
    int b = idx >> 9;
    float z = tau[b] * w1[d] + b1[d];
    out[idx] = z / (1.f + expf(-z));
}

// query += htau[b] + (x_coarse@c_w + c_b) + ((x_t-x_coarse)@d_w + d_b) + id_emb[j]
__global__ __launch_bounds__(256) void query_finalize(float* __restrict__ x,
                                                      const float* __restrict__ htau,
                                                      const float* __restrict__ x_t,
                                                      const float* __restrict__ x_c,
                                                      const float* __restrict__ c_w,
                                                      const float* __restrict__ c_b,
                                                      const float* __restrict__ d_w,
                                                      const float* __restrict__ d_b,
                                                      const float* __restrict__ id_emb) {
    size_t idx = (size_t)blockIdx.x * 256 + threadIdx.x;   // Tt*Dd
    if (idx >= (size_t)Tt * Dd) return;
    int d = (int)(idx & (Dd - 1));
    int t = (int)(idx >> 9);
    int b = t / Jn, j = t % Jn;
    float acc = x[idx] + htau[(size_t)b * Dd + d] + id_emb[(size_t)j * Dd + d] + c_b[d] + d_b[d];
    #pragma unroll
    for (int c = 0; c < 3; ++c) {
        float xc = x_c[(size_t)t * 3 + c];
        float xt = x_t[(size_t)t * 3 + c];
        acc += xc * c_w[c * Dd + d] + (xt - xc) * d_w[c * Dd + d];
    }
    x[idx] = acc;
}

// ---------------- spatial mask (dist > R, with argmin rescue) --------------
__global__ __launch_bounds__(256) void mask_kernel(const float* __restrict__ x_t,
                                                   const float* __restrict__ pc,
                                                   unsigned char* __restrict__ mask) {
    int gw = (blockIdx.x * 256 + threadIdx.x) >> 5;        // wave per (b,j)
    int lane = threadIdx.x & 31;
    if (gw >= Tt) return;
    int b = gw / Jn;
    float qx = x_t[gw * 3 + 0], qy = x_t[gw * 3 + 1], qz = x_t[gw * 3 + 2];
    const float* P = pc + (size_t)b * Nn * 6;
    float dmin = 3.0e38f; int imin = 0;
    for (int n = lane; n < Nn; n += 32) {
        float dx = qx - P[(size_t)n * 6 + 0];
        float dy = qy - P[(size_t)n * 6 + 1];
        float dz = qz - P[(size_t)n * 6 + 2];
        float dist = sqrtf(dx * dx + dy * dy + dz * dz);
        if (dist < dmin) { dmin = dist; imin = n; }
    }
    #pragma unroll
    for (int off = 16; off; off >>= 1) {
        float od = __shfl_xor(dmin, off, 32);
        int   oi = __shfl_xor(imin, off, 32);
        if (od < dmin || (od == dmin && oi < imin)) { dmin = od; imin = oi; }
    }
    bool allfar = dmin > 0.1f;
    for (int n = lane; n < Nn; n += 32) {
        float dx = qx - P[(size_t)n * 6 + 0];
        float dy = qy - P[(size_t)n * 6 + 1];
        float dz = qz - P[(size_t)n * 6 + 2];
        float dist = sqrtf(dx * dx + dy * dy + dz * dz);
        mask[(size_t)gw * Nn + n] = allfar ? (unsigned char)(n != imin)
                                           : (unsigned char)(dist > 0.1f);
    }
}

// ---------------- LayerNorm: wave per 512-wide row -------------------------
__global__ __launch_bounds__(256) void ln_kernel(const float* __restrict__ x,
                                                 const float* __restrict__ g,
                                                 const float* __restrict__ b,
                                                 float* __restrict__ out, int rows) {
    int rw = (blockIdx.x * 256 + threadIdx.x) >> 5;
    int lane = threadIdx.x & 31;
    if (rw >= rows) return;
    const float* r = x + (size_t)rw * Dd;
    float vals[16];
    float s = 0.f;
    #pragma unroll
    for (int i = 0; i < 16; ++i) { vals[i] = r[lane + i * 32]; s += vals[i]; }
    #pragma unroll
    for (int off = 16; off; off >>= 1) s += __shfl_xor(s, off, 32);
    float mean = s * (1.f / Dd);
    float vs = 0.f;
    #pragma unroll
    for (int i = 0; i < 16; ++i) { float d = vals[i] - mean; vs += d * d; }
    #pragma unroll
    for (int off = 16; off; off >>= 1) vs += __shfl_xor(vs, off, 32);
    float inv = rsqrtf(vs * (1.f / Dd) + 1e-5f);
    #pragma unroll
    for (int i = 0; i < 16; ++i) {
        int col = lane + i * 32;
        out[(size_t)rw * Dd + col] = (vals[i] - mean) * inv * g[col] + b[col];
    }
}

// ---------------- self-attention (27x27, topology masked) ------------------
__global__ __launch_bounds__(256) void sattn(const float* __restrict__ qb,
                                             const float* __restrict__ kb,
                                             const float* __restrict__ vb,
                                             float* __restrict__ ob) {
    int b = blockIdx.x >> 3, hh = blockIdx.x & 7;          // block per (b,h)
    __shared__ float S[Jn][Jn + 1];
    __shared__ float Pr[Jn][Jn + 1];
    const size_t base = (size_t)b * Jn * Dd + hh * DH;
    for (int idx = threadIdx.x; idx < Jn * Jn; idx += 256) {
        int qj = idx / Jn, kj = idx % Jn;
        const float* qr = qb + base + (size_t)qj * Dd;
        const float* kr = kb + base + (size_t)kj * Dd;
        float s = 0.f;
        #pragma unroll
        for (int i = 0; i < DH; ++i) s += qr[i] * kr[i];
        bool ok = (qj == kj) || (c_parent[qj] == kj) || (c_parent[kj] == qj);
        S[qj][kj] = s * 0.125f + (ok ? 0.f : NEGBIG);
    }
    __syncthreads();
    if (threadIdx.x < Jn) {
        int j = threadIdx.x;
        float mx = -3.0e38f;
        for (int k = 0; k < Jn; ++k) mx = fmaxf(mx, S[j][k]);
        float sum = 0.f;
        for (int k = 0; k < Jn; ++k) { float e = expf(S[j][k] - mx); Pr[j][k] = e; sum += e; }
        float inv = 1.f / sum;
        for (int k = 0; k < Jn; ++k) Pr[j][k] *= inv;
    }
    __syncthreads();
    for (int idx = threadIdx.x; idx < Jn * DH; idx += 256) {
        int qj = idx >> 6, d = idx & 63;
        float a = 0.f;
        for (int k = 0; k < Jn; ++k) a += Pr[qj][k] * vb[base + (size_t)k * Dd + d];
        ob[base + (size_t)qj * Dd + d] = a;
    }
}

// ---------------- cross-attention: flash-style, wave per (b,h,q) -----------
__global__ __launch_bounds__(864) void cattn(const float* __restrict__ qb,
                                             const _Float16* __restrict__ kbuf,
                                             const _Float16* __restrict__ vbuf,
                                             const unsigned char* __restrict__ maskb,
                                             float* __restrict__ ob) {
    int b = blockIdx.x >> 3, hh = blockIdx.x & 7;          // block per (b,h); 27 waves
    int w = threadIdx.x >> 5;                              // query j
    int lane = threadIdx.x & 31;
    __shared__ float qs[Jn][DH];
    for (int idx = threadIdx.x; idx < Jn * DH; idx += Jn * 32) {
        int j = idx >> 6, d = idx & 63;
        qs[j][d] = qb[((size_t)(b * Jn + j)) * Dd + hh * DH + d];
    }
    __syncthreads();
    const _Float16* Kb = kbuf + ((size_t)b * Nn) * Dd + hh * DH;
    const _Float16* Vb = vbuf + ((size_t)b * Nn) * Dd + hh * DH;
    const unsigned char* mb = maskb + ((size_t)(b * Jn + w)) * Nn;
    float m = -3.0e38f, l = 0.f;
    float accv[DH];
    #pragma unroll
    for (int i = 0; i < DH; ++i) accv[i] = 0.f;
    for (int n = lane; n < Nn; n += 32) {
        const _Float16* kr = Kb + (size_t)n * Dd;
        float s = 0.f;
        #pragma unroll
        for (int i = 0; i < DH; i += 8) {
            v8h kv = *(const v8h*)(kr + i);
            #pragma unroll
            for (int e = 0; e < 8; ++e) s += (float)kv[e] * qs[w][i + e];
        }
        s *= 0.125f;
        if (mb[n]) s += NEGBIG;
        float mn = fmaxf(m, s);
        float c = expf(m - mn);
        float p = expf(s - mn);
        l = l * c + p;
        const _Float16* vr = Vb + (size_t)n * Dd;
        #pragma unroll
        for (int i = 0; i < DH; i += 8) {
            v8h vv = *(const v8h*)(vr + i);
            #pragma unroll
            for (int e = 0; e < 8; ++e) accv[i + e] = accv[i + e] * c + p * (float)vv[e];
        }
        m = mn;
    }
    #pragma unroll
    for (int off = 16; off; off >>= 1) {
        float m2 = __shfl_xor(m, off, 32);
        float l2 = __shfl_xor(l, off, 32);
        float mn = fmaxf(m, m2);
        float c1 = expf(m - mn), c2 = expf(m2 - mn);
        l = l * c1 + l2 * c2;
        #pragma unroll
        for (int i = 0; i < DH; ++i) {
            float o = __shfl_xor(accv[i], off, 32);
            accv[i] = accv[i] * c1 + o * c2;
        }
        m = mn;
    }
    float inv = 1.f / fmaxf(l, 1e-30f);
    for (int i = lane; i < DH; i += 32)
        ob[((size_t)(b * Jn + w)) * Dd + hh * DH + i] = accv[i] * inv;
}

// ---------------- output head: x @ v_w (512x3) + v_b -----------------------
__global__ __launch_bounds__(256) void vout(const float* __restrict__ x,
                                            const float* __restrict__ vw,
                                            const float* __restrict__ vb,
                                            float* __restrict__ out) {
    int idx = blockIdx.x * 256 + threadIdx.x;              // Tt*3
    if (idx >= Tt * 3) return;
    int t = idx / 3, c = idx % 3;
    float a = vb[c];
    const float* xr = x + (size_t)t * Dd;
    for (int d = 0; d < Dd; ++d) a += xr[d] * vw[d * 3 + c];
    out[idx] = a;
}

// ===========================================================================
extern "C" void kernel_launch(void* const* d_in, const int* in_sizes, int n_in,
                              void* d_out, int out_size, void* d_ws, size_t ws_size,
                              hipStream_t stream) {
    // ---- input map (setup_inputs insertion order, depth-first) ----
    const float* x_t      = (const float*)d_in[0];
    const float* tau      = (const float*)d_in[1];
    const float* x_coarse = (const float*)d_in[2];
    const float* pc_raw   = (const float*)d_in[3];
    const float* pc_w = (const float*)d_in[4];
    const float* pc_b = (const float*)d_in[5];
    const float* j_w  = (const float*)d_in[6];
    const float* j_b  = (const float*)d_in[7];
    const float* c_w  = (const float*)d_in[8];
    const float* c_b  = (const float*)d_in[9];
    const float* d_w  = (const float*)d_in[10];
    const float* d_b  = (const float*)d_in[11];
    const float* id_emb = (const float*)d_in[12];
    const float* t_w1 = (const float*)d_in[13];
    const float* t_b1 = (const float*)d_in[14];
    const float* t_w2 = (const float*)d_in[15];
    const float* t_b2 = (const float*)d_in[16];
    const float* dp_w = (const float*)d_in[17];
    const float* dp_b = (const float*)d_in[18];
    const float* v_w  = (const float*)d_in[19];
    const float* v_b  = (const float*)d_in[20];
    const int LBASE = 21, LSTR = 26;

    // ---- workspace carve-up ----
    size_t off = 0;
    auto alloc = [&](size_t bytes) -> void* {
        void* p = (char*)d_ws + off;
        off = (off + bytes + 255) & ~(size_t)255;
        return p;
    };
    _Float16* t_w2t = (_Float16*)alloc((size_t)Dd * Dd * 2);
    _Float16* dp_wt = (_Float16*)alloc((size_t)Dd * 2 * Dd * 2);
    _Float16* lw_t[6][10];   // sa q/k/v/o, ca q/k/v/o, ff1, ff2
    for (int l = 0; l < 6; ++l) {
        for (int m = 0; m < 8; ++m) lw_t[l][m] = (_Float16*)alloc((size_t)Dd * Dd * 2);
        lw_t[l][8] = (_Float16*)alloc((size_t)Dd * FFd * 2);  // ff1t [2048][512]
        lw_t[l][9] = (_Float16*)alloc((size_t)FFd * Dd * 2);  // ff2t [512][2048]
    }
    _Float16* mem_h = (_Float16*)alloc((size_t)Bsz * Nn * Dd * 2);
    _Float16* kbuf  = (_Float16*)alloc((size_t)Bsz * Nn * Dd * 2);
    _Float16* vbuf  = (_Float16*)alloc((size_t)Bsz * Nn * Dd * 2);
    float* x    = (float*)alloc((size_t)Tt * Dd * 4);
    float* h    = (float*)alloc((size_t)Tt * Dd * 4);
    float* qb   = (float*)alloc((size_t)Tt * Dd * 4);
    float* kb2  = (float*)alloc((size_t)Tt * Dd * 4);
    float* vb2  = (float*)alloc((size_t)Tt * Dd * 4);
    float* ao   = (float*)alloc((size_t)Tt * Dd * 4);
    float* cc   = (float*)alloc((size_t)Tt * 2 * Dd * 4);
    float* hxt  = (float*)alloc((size_t)Tt * Dd * 4);
    float* t1   = (float*)alloc((size_t)Bsz * Dd * 4);
    float* htau = (float*)alloc((size_t)Bsz * Dd * 4);
    float* ffb  = (float*)alloc((size_t)Tt * FFd * 4);
    unsigned char* maskb = (unsigned char*)alloc((size_t)Tt * Nn);

    auto conv = [&](const float* w, _Float16* wt, int K, int N) {
        size_t total = (size_t)K * N;
        wconv<<<(unsigned)((total + 255) / 256), 256, 0, stream>>>(w, wt, K, N);
    };

    // ---- weight transpose/convert (L2-resident f16 weights) ----
    conv(t_w2, t_w2t, Dd, Dd);
    conv(dp_w, dp_wt, 2 * Dd, Dd);
    for (int l = 0; l < 6; ++l) {
        int base = LBASE + l * LSTR;
        const int wi[8] = {0, 2, 4, 6, 8, 10, 12, 14};      // sa wq/wk/wv/wo, ca wq/wk/wv/wo
        for (int m = 0; m < 8; ++m) conv((const float*)d_in[base + wi[m]], lw_t[l][m], Dd, Dd);
        conv((const float*)d_in[base + 22], lw_t[l][8], Dd, FFd);   // ff_w1
        conv((const float*)d_in[base + 24], lw_t[l][9], FFd, Dd);   // ff_w2
    }

    // ---- query path + memory embedding + mask ----
    {
        size_t tot = (size_t)Bsz * Nn * Dd;
        mem_embed<<<(unsigned)((tot + 255) / 256), 256, 0, stream>>>(pc_raw, pc_w, pc_b, mem_h);
    }
    {
        unsigned g = (Tt * Dd + 255) / 256;
        proj3<<<g, 256, 0, stream>>>(x_t, j_w, j_b, hxt);
        concat_parent<<<g, 256, 0, stream>>>(hxt, cc);
    }
    gemm_wmma<false, false, false, false><<<gemm_grid(Tt, Dd), 256, gemm_lds(2 * Dd), stream>>>(
        cc, dp_wt, dp_b, x, nullptr, Tt, Dd, 2 * Dd);
    tau_silu<<<(Bsz * Dd + 255) / 256, 256, 0, stream>>>(tau, t_w1, t_b1, t1);
    gemm_wmma<false, false, false, false><<<gemm_grid(Bsz, Dd), 256, gemm_lds(Dd), stream>>>(
        t1, t_w2t, t_b2, htau, nullptr, Bsz, Dd, Dd);
    query_finalize<<<(Tt * Dd + 255) / 256, 256, 0, stream>>>(
        x, htau, x_t, x_coarse, c_w, c_b, d_w, d_b, id_emb);
    mask_kernel<<<(Tt * 32 + 255) / 256, 256, 0, stream>>>(x_t, pc_raw, maskb);

    // ---- transformer layers ----
    for (int l = 0; l < 6; ++l) {
        int base = LBASE + l * LSTR;
        const float* sa_bq = (const float*)d_in[base + 1];
        const float* sa_bk = (const float*)d_in[base + 3];
        const float* sa_bv = (const float*)d_in[base + 5];
        const float* sa_bo = (const float*)d_in[base + 7];
        const float* ca_bq = (const float*)d_in[base + 9];
        const float* ca_bk = (const float*)d_in[base + 11];
        const float* ca_bv = (const float*)d_in[base + 13];
        const float* ca_bo = (const float*)d_in[base + 15];
        const float* ln1_g = (const float*)d_in[base + 16];
        const float* ln1_b = (const float*)d_in[base + 17];
        const float* ln2_g = (const float*)d_in[base + 18];
        const float* ln2_b = (const float*)d_in[base + 19];
        const float* ln3_g = (const float*)d_in[base + 20];
        const float* ln3_b = (const float*)d_in[base + 21];
        const float* ff_b1 = (const float*)d_in[base + 23];
        const float* ff_b2 = (const float*)d_in[base + 25];
        const unsigned lnGrid = (Tt * 32 + 255) / 256;

        // self-attention block
        ln_kernel<<<lnGrid, 256, 0, stream>>>(x, ln1_g, ln1_b, h, Tt);
        gemm_wmma<false, false, false, false><<<gemm_grid(Tt, Dd), 256, gemm_lds(Dd), stream>>>(
            h, lw_t[l][0], sa_bq, qb, nullptr, Tt, Dd, Dd);
        gemm_wmma<false, false, false, false><<<gemm_grid(Tt, Dd), 256, gemm_lds(Dd), stream>>>(
            h, lw_t[l][1], sa_bk, kb2, nullptr, Tt, Dd, Dd);
        gemm_wmma<false, false, false, false><<<gemm_grid(Tt, Dd), 256, gemm_lds(Dd), stream>>>(
            h, lw_t[l][2], sa_bv, vb2, nullptr, Tt, Dd, Dd);
        sattn<<<Bsz * Hh, 256, 0, stream>>>(qb, kb2, vb2, ao);
        gemm_wmma<false, false, false, true><<<gemm_grid(Tt, Dd), 256, gemm_lds(Dd), stream>>>(
            ao, lw_t[l][3], sa_bo, x, x, Tt, Dd, Dd);

        // cross-attention block (K/V projections = dominant GEMMs, M=65536)
        ln_kernel<<<lnGrid, 256, 0, stream>>>(x, ln2_g, ln2_b, h, Tt);
        gemm_wmma<false, false, false, false><<<gemm_grid(Tt, Dd), 256, gemm_lds(Dd), stream>>>(
            h, lw_t[l][4], ca_bq, qb, nullptr, Tt, Dd, Dd);
        gemm_wmma<true, true, false, false><<<gemm_grid(Bsz * Nn, Dd), 256, gemm_lds(Dd), stream>>>(
            mem_h, lw_t[l][5], ca_bk, kbuf, nullptr, Bsz * Nn, Dd, Dd);
        gemm_wmma<true, true, false, false><<<gemm_grid(Bsz * Nn, Dd), 256, gemm_lds(Dd), stream>>>(
            mem_h, lw_t[l][6], ca_bv, vbuf, nullptr, Bsz * Nn, Dd, Dd);
        cattn<<<Bsz * Hh, Jn * 32, 0, stream>>>(qb, kbuf, vbuf, maskb, ao);
        gemm_wmma<false, false, false, true><<<gemm_grid(Tt, Dd), 256, gemm_lds(Dd), stream>>>(
            ao, lw_t[l][7], ca_bo, x, x, Tt, Dd, Dd);

        // feed-forward block
        ln_kernel<<<lnGrid, 256, 0, stream>>>(x, ln3_g, ln3_b, h, Tt);
        gemm_wmma<false, false, true, false><<<gemm_grid(Tt, FFd), 256, gemm_lds(Dd), stream>>>(
            h, lw_t[l][8], ff_b1, ffb, nullptr, Tt, FFd, Dd);
        gemm_wmma<false, false, false, true><<<gemm_grid(Tt, Dd), 256, gemm_lds(FFd), stream>>>(
            ffb, lw_t[l][9], ff_b2, x, x, Tt, Dd, FFd);
    }

    // ---- output head ----
    vout<<<(Tt * 3 + 255) / 256, 256, 0, stream>>>(x, v_w, v_b, (float*)d_out);
}
